// AutoEncoder_14328010899794
// MI455X (gfx1250) — matrix operand order repaired and checked by reading, not compiled, not run on hardware
//
#include <hip/hip_runtime.h>
#include <hip/hip_bf16.h>
#include <math.h>

// ---------------------------------------------------------------------------
// Set-autoencoder forward for MI455X (gfx1250, wave32, WMMA).
// DIM=256, HID=512, MAXN=16, B=16384, V_MID=384, D_MID=384, K_MID=264.
// ---------------------------------------------------------------------------

typedef __attribute__((ext_vector_type(16))) _Float16 v16h;
typedef __attribute__((ext_vector_type(8)))  float    v8f;

#define DIM   256
#define HID   512
#define MAXN  16
#define NBATCH 16384
#define KMID  264
#define VMID  384
#define DMID  384

union FragH {
    v16h h;
    unsigned int u[8];
    uint4 q[2];
};

__device__ __forceinline__ float mishf(float x) {
    float sp = (x > 20.0f) ? x : log1pf(__expf(x));
    return x * tanhf(sp);
}

__device__ __forceinline__ unsigned pack2h(float x, float y) {
    _Float16 h0 = (_Float16)x;
    _Float16 h1 = (_Float16)y;
    unsigned short s0 = __builtin_bit_cast(unsigned short, h0);
    unsigned short s1 = __builtin_bit_cast(unsigned short, h1);
    return (unsigned)s0 | ((unsigned)s1 << 16);
}

// ---------------------------------------------------------------------------
// Weight pre-pack: f32 row-major [K][NOUT] -> f16 WMMA B-fragment layout
//   packed[((ks*CT + ct)*32 + lane)*16 + e] = W[(ks*32 + 8*(lane>>4) + (e&7) + 16*(e>>3))*NOUT
//                                               + ct*16 + (lane&15)]
// ---------------------------------------------------------------------------
__global__ void pack_w_kernel(const float* __restrict__ W,
                              _Float16* __restrict__ Wp, int K, int NOUT) {
    int o = blockIdx.x * blockDim.x + threadIdx.x;
    if (o >= K * NOUT) return;
    int e    = o & 15;
    int lane = (o >> 4) & 31;
    int rest = o >> 9;
    int CT   = NOUT >> 4;
    int ct   = rest % CT;
    int ks   = rest / CT;
    int k = ks * 32 + ((lane >> 4) << 3) + (e & 7) + ((e >> 3) << 4);
    int n = ct * 16 + (lane & 15);
    Wp[o] = (_Float16)W[(size_t)k * NOUT + n];
}

// ---------------------------------------------------------------------------
// Segment bookkeeping: batch is sorted, every segment non-empty.
// ---------------------------------------------------------------------------
__global__ void seg_starts_kernel(const int* __restrict__ batch,
                                  int* __restrict__ starts, int N, int B) {
    int i = blockIdx.x * blockDim.x + threadIdx.x;
    if (i < N) {
        int b = batch[i];
        if (i == 0 || batch[i - 1] != b) starts[b] = i;
    }
    if (i == 0) starts[B] = N;
}

__global__ void seg_kpos_kernel(const int* __restrict__ batch,
                                const int* __restrict__ starts,
                                int* __restrict__ kpos, int N) {
    int i = blockIdx.x * blockDim.x + threadIdx.x;
    if (i < N) kpos[i] = i - starts[batch[i]];
}

// z[b][c] = n[b]*card_W[c] + card_b[c]   (cardinality encoding)
__global__ void z_init_kernel(const int* __restrict__ starts,
                              const float* __restrict__ cW,
                              const float* __restrict__ cb,
                              float* __restrict__ z, int B) {
    int o = blockIdx.x * blockDim.x + threadIdx.x;
    if (o >= B * HID) return;
    int b = o >> 9;
    int c = o & 511;
    float n = (float)(starts[b + 1] - starts[b]);
    z[o] = n * cW[c] + cb[c];
}

// ---------------------------------------------------------------------------
// q_table[16][512] = mlp_ln(one_hot(k), key_net). Single workgroup.
// ---------------------------------------------------------------------------
__global__ void qtable_kernel(const float* __restrict__ kW1,
                              const float* __restrict__ kb1,
                              const float* __restrict__ kg,
                              const float* __restrict__ kbt,
                              const float* __restrict__ kW2,
                              const float* __restrict__ kb2,
                              float* __restrict__ qt) {
    __shared__ float h[MAXN * KMID];
    __shared__ float stats[2 * MAXN];
    for (int o = threadIdx.x; o < MAXN * KMID; o += blockDim.x) {
        int c = o % KMID;
        h[o] = kW1[o] + kb1[c];
    }
    __syncthreads();
    if (threadIdx.x < MAXN) {
        int r = threadIdx.x;
        float s = 0.f, q = 0.f;
        for (int c = 0; c < KMID; ++c) {
            float v = h[r * KMID + c];
            s += v; q += v * v;
        }
        float mean = s / KMID;
        float var  = q / KMID - mean * mean;
        stats[r] = mean;
        stats[MAXN + r] = rsqrtf(var + 1e-5f);
    }
    __syncthreads();
    for (int o = threadIdx.x; o < MAXN * KMID; o += blockDim.x) {
        int r = o / KMID, c = o % KMID;
        float v = (h[o] - stats[r]) * stats[MAXN + r] * kg[c] + kbt[c];
        h[o] = mishf(v);
    }
    __syncthreads();
    for (int o = threadIdx.x; o < MAXN * HID; o += blockDim.x) {
        int r = o >> 9, c = o & 511;
        float s = kb2[c];
        for (int j = 0; j < KMID; ++j) s += h[r * KMID + j] * kW2[(size_t)j * HID + c];
        qt[o] = s;
    }
}

// z[b][c] += sum_{i in segment b} Y[i][c]   (fixed order -> deterministic)
__global__ void seg_sum_kernel(const _Float16* __restrict__ Y,
                               const int* __restrict__ starts,
                               float* __restrict__ z, int B) {
    int o = blockIdx.x * blockDim.x + threadIdx.x;
    if (o >= B * HID) return;
    int b = o >> 9;
    int c = o & 511;
    float acc = z[o];
    int i0 = starts[b], i1 = starts[b + 1];
    for (int i = i0; i < i1; ++i) acc += (float)Y[(size_t)i * HID + c];
    z[o] = acc;
}

__global__ void out_batch_kernel(const int* __restrict__ batch,
                                 float* __restrict__ out, int N) {
    int i = blockIdx.x * blockDim.x + threadIdx.x;
    if (i < N) out[i] = (float)batch[i];
}

// ---------------------------------------------------------------------------
// Tiled WMMA GEMM: one workgroup = 32 rows x NOUT cols; 8 waves.
// The full 32 x K A panel is staged into LDS ONCE (one barrier), then the
// K-loop is pure ds_load_b128 + global_load_b128 + v_wmma (pipelineable,
// no barriers). Each wave owns 2 row tiles x NT col tiles: every B fragment
// feeds 2 WMMAs, every A fragment feeds NT WMMAs.
//   ASRC : 0 = f16 activations, 1 = f32 x (convert), 2 = z[batch]*q[kpos]
//   LN   : fuse LayerNorm over NOUT via LDS tile (reuses A panel LDS)
//   MISH : fuse mish
//   OMODE: 0 = f16 store, 1 = f16 store of (v*q[kpos][col]), 2 = f32 store
// ---------------------------------------------------------------------------
template <int K, int NOUT, int ASRC, bool LN, bool MISH, int OMODE>
__global__ __launch_bounds__(256) void gemm_tile(
    const void* __restrict__ A, const _Float16* __restrict__ Wp,
    const float* __restrict__ bias, const float* __restrict__ gamma,
    const float* __restrict__ beta, const float* __restrict__ zsrc,
    const float* __restrict__ qtab, const int* __restrict__ batch,
    const int* __restrict__ kpos, void* __restrict__ out) {
    constexpr int CT = NOUT / 16;      // col tiles
    constexpr int NT = CT / 8;         // col tiles per wave (8 waves)
    constexpr int ADW = 16 * K;        // 32 rows * K halves / 2 dwords
    constexpr int LNDW = LN ? 32 * NOUT : 0;
    constexpr int SHDW = (ADW > LNDW) ? ADW : LNDW;

    __shared__ __align__(16) unsigned shbuf[SHDW];   // A panel, later LN tile
    unsigned* sA32 = shbuf;
    float* sC = (float*)shbuf;

    const int tid  = threadIdx.x;
    const int lane = tid & 31;
    const int wave = tid >> 5;
    const int hi   = lane >> 4;
    const int m    = lane & 15;
    const int row0 = blockIdx.x << 5;  // 32 rows per block

    // ---- stage the entire 32 x K A panel (f16) into LDS, coalesced ----
    for (int d = tid; d < ADW; d += 256) {
        const int hidx = d << 1;
        const int r    = hidx / K;
        const int c    = hidx % K;       // even
        const int row  = row0 + r;
        unsigned u;
        if constexpr (ASRC == 0) {
            const unsigned* Af = (const unsigned*)A;
            u = Af[((size_t)row * K + c) >> 1];
        } else if constexpr (ASRC == 1) {
            const float2* Af = (const float2*)A;
            float2 f = Af[((size_t)row * K + c) >> 1];
            u = pack2h(f.x, f.y);
        } else {
            const int b  = batch[row];
            const int kp = kpos[row];
            float v0 = zsrc[(size_t)b * HID + c]     * qtab[kp * HID + c];
            float v1 = zsrc[(size_t)b * HID + c + 1] * qtab[kp * HID + c + 1];
            u = pack2h(v0, v1);
        }
        sA32[d] = u;
    }
    __syncthreads();

    v8f acc[2][NT] = {};

    for (int kk = 0; kk < K; kk += 32) {
        // A fragments for both row tiles (ISA 7.12.2: 16-bit A 16x32 layout).
        // Two 16B-aligned uint4 LDS loads per fragment -> ds_load_b128 into
        // contiguous VGPR quads (no scatter/mov chains before v_wmma).
        FragH a[2];
#pragma unroll
        for (int rt = 0; rt < 2; ++rt) {
            const uint4* pa =
                (const uint4*)&sA32[((rt * 16 + m) * K + kk + hi * 8) >> 1];
            a[rt].q[0] = pa[0];   // halves kk+8*hi   .. +7
            a[rt].q[1] = pa[2];   // halves kk+16+8*hi .. +7
        }
        const int ks = kk >> 5;
#pragma unroll
        for (int j = 0; j < NT; ++j) {
            const int ct = wave * NT + j;
            const uint4* pb =
                (const uint4*)Wp + (((size_t)ks * CT + ct) * 32 + lane) * 2;
            FragH b;
            b.q[0] = pb[0];
            b.q[1] = pb[1];
            acc[0][j] = __builtin_amdgcn_wmma_f32_16x16x32_f16(
                false, a[0].h, false, b.h, (short)0, acc[0][j], false, false);
            acc[1][j] = __builtin_amdgcn_wmma_f32_16x16x32_f16(
                false, a[1].h, false, b.h, (short)0, acc[1][j], false, false);
        }
    }

    if constexpr (LN) {
        __syncthreads();   // A panel dead; reuse LDS as 32 x NOUT f32 tile
#pragma unroll
        for (int rt = 0; rt < 2; ++rt) {
#pragma unroll
            for (int j = 0; j < NT; ++j) {
                const int col = (wave * NT + j) * 16 + m;
#pragma unroll
                for (int r = 0; r < 8; ++r) {
                    const int lr = rt * 16 + r + 8 * hi;
                    sC[lr * NOUT + col] = acc[rt][j][r] + bias[col];
                }
            }
        }
        __syncthreads();
        const int lr = tid >> 3;   // 32 rows, 8 threads per row
        const int li = tid & 7;
        float s = 0.f, q = 0.f;
        for (int c = li; c < NOUT; c += 8) {
            const float v = sC[lr * NOUT + c];
            s += v; q += v * v;
        }
#pragma unroll
        for (int msk = 4; msk >= 1; msk >>= 1) {
            s += __shfl_xor(s, msk, 8);
            q += __shfl_xor(q, msk, 8);
        }
        const float mean = s / NOUT;
        const float var  = q / NOUT - mean * mean;
        const float rstd = rsqrtf(var + 1e-5f);
        _Float16* O = (_Float16*)out;
        for (int c = li; c < NOUT; c += 8) {
            float v = (sC[lr * NOUT + c] - mean) * rstd * gamma[c] + beta[c];
            if constexpr (MISH) v = mishf(v);
            O[(size_t)(row0 + lr) * NOUT + c] = (_Float16)v;
        }
    } else {
#pragma unroll
        for (int rt = 0; rt < 2; ++rt) {
#pragma unroll
            for (int r = 0; r < 8; ++r) {
                const int grow = row0 + rt * 16 + r + 8 * hi;
                int kp = 0;
                if constexpr (OMODE == 1) kp = kpos[grow];
#pragma unroll
                for (int j = 0; j < NT; ++j) {
                    const int col = (wave * NT + j) * 16 + m;
                    float v = acc[rt][j][r] + bias[col];
                    if constexpr (MISH) v = mishf(v);
                    if constexpr (OMODE == 0) {
                        ((_Float16*)out)[(size_t)grow * NOUT + col] = (_Float16)v;
                    } else if constexpr (OMODE == 1) {
                        v *= qtab[kp * HID + col];
                        ((_Float16*)out)[(size_t)grow * NOUT + col] = (_Float16)v;
                    } else {
                        ((float*)out)[(size_t)grow * NOUT + col] = v;
                    }
                }
            }
        }
    }
}

// ---------------------------------------------------------------------------
extern "C" void kernel_launch(void* const* d_in, const int* in_sizes, int n_in,
                              void* d_out, int out_size, void* d_ws, size_t ws_size,
                              hipStream_t stream) {
    const float* x      = (const float*)d_in[0];
    const int*   batch  = (const int*)d_in[1];
    const float* kW1    = (const float*)d_in[3];
    const float* kb1    = (const float*)d_in[4];
    const float* kg     = (const float*)d_in[5];
    const float* kbt    = (const float*)d_in[6];
    const float* kW2    = (const float*)d_in[7];
    const float* kb2    = (const float*)d_in[8];
    const float* vW1    = (const float*)d_in[9];
    const float* vb1    = (const float*)d_in[10];
    const float* vg     = (const float*)d_in[11];
    const float* vbt    = (const float*)d_in[12];
    const float* vW2    = (const float*)d_in[13];
    const float* vb2    = (const float*)d_in[14];
    const float* dW1    = (const float*)d_in[15];
    const float* db1    = (const float*)d_in[16];
    const float* dW2    = (const float*)d_in[17];
    const float* db2    = (const float*)d_in[18];
    const float* card_W = (const float*)d_in[21];
    const float* card_b = (const float*)d_in[22];

    const int N = in_sizes[1];     // 139264 (multiple of 32)
    const int B = NBATCH;          // reference constant

    char* w = (char*)d_ws;
    auto alloc = [&](size_t bytes) -> void* {
        void* p = (void*)w;
        w += (bytes + 255) & ~(size_t)255;
        return p;
    };
    _Float16* vW1p = (_Float16*)alloc((size_t)DIM * VMID * 2);
    _Float16* vW2p = (_Float16*)alloc((size_t)VMID * HID * 2);
    _Float16* dW1p = (_Float16*)alloc((size_t)HID * DMID * 2);
    _Float16* dW2p = (_Float16*)alloc((size_t)DMID * DIM * 2);
    float*    qt   = (float*)alloc((size_t)MAXN * HID * 4);
    int*      sts  = (int*)alloc((size_t)(B + 1) * 4);
    int*      kpos = (int*)alloc((size_t)N * 4);
    float*    z    = (float*)alloc((size_t)B * HID * 4);
    _Float16* H1   = (_Float16*)alloc((size_t)N * VMID * 2);  // reused as D1
    _Float16* Y    = (_Float16*)alloc((size_t)N * HID * 2);

    const int TPB = 256;
    pack_w_kernel<<<(DIM * VMID + TPB - 1) / TPB, TPB, 0, stream>>>(vW1, vW1p, DIM, VMID);
    pack_w_kernel<<<(VMID * HID + TPB - 1) / TPB, TPB, 0, stream>>>(vW2, vW2p, VMID, HID);
    pack_w_kernel<<<(HID * DMID + TPB - 1) / TPB, TPB, 0, stream>>>(dW1, dW1p, HID, DMID);
    pack_w_kernel<<<(DMID * DIM + TPB - 1) / TPB, TPB, 0, stream>>>(dW2, dW2p, DMID, DIM);
    seg_starts_kernel<<<(N + TPB - 1) / TPB, TPB, 0, stream>>>(batch, sts, N, B);
    seg_kpos_kernel<<<(N + TPB - 1) / TPB, TPB, 0, stream>>>(batch, sts, kpos, N);
    z_init_kernel<<<(B * HID + TPB - 1) / TPB, TPB, 0, stream>>>(sts, card_W, card_b, z, B);
    qtable_kernel<<<1, TPB, 0, stream>>>(kW1, kb1, kg, kbt, kW2, kb2, qt);

    const int RB = N / 32;   // 32-row blocks
    // GEMM1: x(f32)[N,256] @ vW1 -> LN -> mish -> H1 f16 [N,384]
    gemm_tile<DIM, VMID, 1, true, true, 0><<<RB, TPB, 0, stream>>>(
        x, vW1p, vb1, vg, vbt, nullptr, nullptr, nullptr, nullptr, H1);
    // GEMM2: H1 @ vW2 + vb2, *q[kpos] -> Y f16 [N,512]
    gemm_tile<VMID, HID, 0, false, false, 1><<<RB, TPB, 0, stream>>>(
        H1, vW2p, vb2, nullptr, nullptr, nullptr, qt, nullptr, kpos, Y);
    // deterministic segment sum into z (z already holds cardinality encoding)
    seg_sum_kernel<<<(B * HID + TPB - 1) / TPB, TPB, 0, stream>>>(Y, sts, z, B);
    // GEMM3: (z[batch]*q[kpos]) @ dW1 + db1 -> mish -> D1 f16 [N,384] (reuse H1)
    gemm_tile<HID, DMID, 2, false, true, 0><<<RB, TPB, 0, stream>>>(
        nullptr, dW1p, db1, nullptr, nullptr, z, qt, batch, kpos, H1);
    // GEMM4: D1 @ dW2 + db2 -> xr f32 [N,256] into d_out
    gemm_tile<DMID, DIM, 0, false, false, 2><<<RB, TPB, 0, stream>>>(
        H1, dW2p, db2, nullptr, nullptr, nullptr, nullptr, nullptr, nullptr, d_out);
    // second tuple output: batch as float
    out_batch_kernel<<<(N + TPB - 1) / TPB, TPB, 0, stream>>>(
        batch, (float*)d_out + (size_t)N * DIM, N);
}
